// TI_GATNoEdgeAttrs_59365037966013
// MI455X (gfx1250) — matrix operand order repaired and checked by reading, not compiled, not run on hardware
//
#include <hip/hip_runtime.h>
#include <hip/hip_bf16.h>

#define HEADS 4
#define CDIM 16
#define HID 64
#define FIN 128
#define NLAYERS 6
#define NEG_SLOPE 0.2f

typedef __attribute__((ext_vector_type(16))) _Float16 v16h;
typedef __attribute__((ext_vector_type(8)))  _Float16 v8h;
typedef __attribute__((ext_vector_type(8)))  float    v8f;

__device__ __forceinline__ float lrelu(float x) {
  return x > 0.0f ? x : NEG_SLOPE * x;
}

// ---------------------------------------------------------------------------
// Small utility kernels
// ---------------------------------------------------------------------------
__global__ void zero_i32(int* __restrict__ p, int n) {
  int i = blockIdx.x * blockDim.x + threadIdx.x;
  if (i < n) p[i] = 0;
}

__global__ void copy_i32(const int* __restrict__ a, int* __restrict__ b, int n) {
  int i = blockIdx.x * blockDim.x + threadIdx.x;
  if (i < n) b[i] = a[i];
}

// Split-f16 weight conversion: W [K,64] fp32 row-major -> Wt_hi/Wt_lo [64,K] f16
// (transposed so each WMMA B-lane reads two runs of 8 contiguous halves, at k and k+16 = two b128 loads).
// w = hi + lo + O(2^-22): hi = rn_f16(w), lo = rn_f16(w - hi).
__global__ void wcvt(const float* __restrict__ W, _Float16* __restrict__ Wt_hi,
                     _Float16* __restrict__ Wt_lo, int K) {
  int idx = blockIdx.x * blockDim.x + threadIdx.x;
  if (idx < K * HID) {
    int k = idx / HID, n = idx % HID;
    float w = W[idx];
    _Float16 hi = (_Float16)w;
    _Float16 lo = (_Float16)(w - (float)hi);
    Wt_hi[n * K + k] = hi;
    Wt_lo[n * K + k] = lo;
  }
}

// ---------------------------------------------------------------------------
// CSR build (dst-sorted adjacency, self-loops appended implicitly)
// ---------------------------------------------------------------------------
__global__ void edge_count(const int* __restrict__ dst, int* __restrict__ cnt,
                           int E, int N) {
  int e = blockIdx.x * blockDim.x + threadIdx.x;
  if (e >= E + N) return;
  int d = (e < E) ? dst[e] : (e - E);
  atomicAdd(&cnt[d], 1);
}

// Single-block exclusive scan (one-time cost; ~100k elements, 1024 threads).
__global__ void scan_excl(const int* __restrict__ counts, int* __restrict__ row_ptr, int n) {
  __shared__ int buf[1024];
  __shared__ int carry;
  const int tid = threadIdx.x;
  if (tid == 0) carry = 0;
  __syncthreads();
  for (int base = 0; base < n; base += 1024) {
    int i = base + tid;
    int v = (i < n) ? counts[i] : 0;
    buf[tid] = v;
    __syncthreads();
    for (int off = 1; off < 1024; off <<= 1) {
      int t = (tid >= off) ? buf[tid - off] : 0;
      __syncthreads();
      buf[tid] += t;
      __syncthreads();
    }
    if (i < n) row_ptr[i] = carry + buf[tid] - v;
    __syncthreads();
    if (tid == 0) carry += buf[1023];
    __syncthreads();
  }
  if (tid == 0) row_ptr[n] = carry;
}

__global__ void edge_scatter(const int* __restrict__ src, const int* __restrict__ dst,
                             int* __restrict__ cursor, int* __restrict__ col,
                             int E, int N) {
  int e = blockIdx.x * blockDim.x + threadIdx.x;
  if (e >= E + N) return;
  int s, d;
  if (e < E) { s = src[e]; d = dst[e]; } else { s = d = e - E; }
  int pos = atomicAdd(&cursor[d], 1);
  col[pos] = s;
}

// ---------------------------------------------------------------------------
// Split-f16 WMMA GEMM:  H[n,0:64] = A[n,0:K] @ W[K,64], fp32-equivalent via
// acc += a_hi*w_hi + a_lo*w_hi + a_hi*w_lo  (all fp32 accumulate).
// One wave per 16-row tile; 4 column tiles cover N=64 (one per head).
// ---------------------------------------------------------------------------
__global__ void __launch_bounds__(256)
gemm_wmma_split(const float* __restrict__ A, const _Float16* __restrict__ Bhi,
                const _Float16* __restrict__ Blo, float* __restrict__ H,
                int nrows, int K) {
  const int lane = threadIdx.x & 31;
  const int wave = threadIdx.x >> 5;
  const int tile = blockIdx.x * 8 + wave;
  if (tile * 16 >= nrows) return;            // tile-uniform: EXEC stays all-ones
  const int r  = lane & 15;
  const int hi = lane >> 4;                  // half-wave select
  const int row = tile * 16 + r;
  const int ka = hi ? 8 : 0;                 // A frag: lanes 16-31 hold K+8.. / +24..
  const int kb = hi ? 8 : 0;                 // B frag: lanes 16-31 hold K+8.. / +24..
  v8f acc0 = {}, acc1 = {}, acc2 = {}, acc3 = {};
  const float* arow = A + (size_t)row * K;
  const size_t boff = (size_t)r * K;
  for (int k0 = 0; k0 < K; k0 += 32) {
    // ---- A fragment: split into hi + residual lo ---------------------------
    float af[16];
    const float4* p = (const float4*)(arow + k0 + ka);
    float4 p0 = p[0], p1 = p[1];
    const float4* q = (const float4*)(arow + k0 + ka + 16);
    float4 q0 = q[0], q1 = q[1];
    *(float4*)(af + 0)  = p0;  *(float4*)(af + 4)  = p1;
    *(float4*)(af + 8)  = q0;  *(float4*)(af + 12) = q1;
    v16h ah, al;
#pragma unroll
    for (int i = 0; i < 16; ++i) {
      _Float16 h16 = (_Float16)af[i];
      ah[i] = h16;
      al[i] = (_Float16)(af[i] - (float)h16);
    }
#pragma unroll
    for (int j = 0; j < 4; ++j) {
      const _Float16* bh = Bhi + boff + (size_t)j * 16 * K + k0 + kb;
      const _Float16* bl = Blo + boff + (size_t)j * 16 * K + k0 + kb;
      v8h h0 = *(const v8h*)(bh);  v8h h1 = *(const v8h*)(bh + 16);
      v8h l0 = *(const v8h*)(bl);  v8h l1 = *(const v8h*)(bl + 16);
      v16h bhv = __builtin_shufflevector(h0, h1, 0,1,2,3,4,5,6,7,8,9,10,11,12,13,14,15);
      v16h blv = __builtin_shufflevector(l0, l1, 0,1,2,3,4,5,6,7,8,9,10,11,12,13,14,15);
      v8f acc = j == 0 ? acc0 : j == 1 ? acc1 : j == 2 ? acc2 : acc3;
      acc = __builtin_amdgcn_wmma_f32_16x16x32_f16(false, ah, false, bhv, (short)0, acc, false, false);
      acc = __builtin_amdgcn_wmma_f32_16x16x32_f16(false, al, false, bhv, (short)0, acc, false, false);
      acc = __builtin_amdgcn_wmma_f32_16x16x32_f16(false, ah, false, blv, (short)0, acc, false, false);
      if (j == 0) acc0 = acc; else if (j == 1) acc1 = acc; else if (j == 2) acc2 = acc; else acc3 = acc;
    }
  }
  // C/D layout: VGPR v, lanes 0-15 -> (M=v, N=lane); lanes 16-31 -> (M=v+8)
  float* orow = H + (size_t)tile * 16 * HID;
  const int m0 = hi ? 8 : 0;
#pragma unroll
  for (int v = 0; v < 8; ++v) {
    const int m = m0 + v;
    orow[m * HID +      r] = acc0[v];
    orow[m * HID + 16 + r] = acc1[v];
    orow[m * HID + 32 + r] = acc2[v];
    orow[m * HID + 48 + r] = acc3[v];
  }
}

// ---------------------------------------------------------------------------
// Per-node attention logits: alpha_s[n,h] = <h[n,h,:], a_src[h,:]>, same for dst
// ---------------------------------------------------------------------------
__global__ void alpha_k(const float* __restrict__ h, const float* __restrict__ asrc,
                        const float* __restrict__ adst, float* __restrict__ as_,
                        float* __restrict__ ad_, int N) {
  int idx = blockIdx.x * blockDim.x + threadIdx.x;
  if (idx >= N * HEADS) return;
  int n = idx >> 2, hd = idx & 3;
  const float* hp = h + (size_t)n * HID + hd * CDIM;
  const float* a1 = asrc + hd * CDIM;
  const float* a2 = adst + hd * CDIM;
  float s = 0.0f, d = 0.0f;
#pragma unroll
  for (int c = 0; c < CDIM; ++c) {
    float v = hp[c];
    s = fmaf(v, a1[c], s);
    d = fmaf(v, a2[c], d);
  }
  as_[idx] = s;
  ad_[idx] = d;
}

// ---------------------------------------------------------------------------
// Gather-based softmax aggregation: one wave32 per destination node.
// Single online-softmax pass (fused max+denominator, shfl_xor merge), then
// each lane owns 2 output channels; edge ids are loaded coalesced in groups
// of 32, their h[src] rows prefetched, then broadcast from registers via shfl.
// ---------------------------------------------------------------------------
__global__ void __launch_bounds__(256)
gat_aggregate(const float* __restrict__ h, const float* __restrict__ as_,
              const float* __restrict__ ad_, const float* __restrict__ bias,
              const int* __restrict__ row_ptr, const int* __restrict__ col,
              float* __restrict__ out, int N) {
  const int wid  = (blockIdx.x * blockDim.x + threadIdx.x) >> 5;
  const int lane = threadIdx.x & 31;
  if (wid >= N) return;
  const int s0 = row_ptr[wid];
  const int s1 = row_ptr[wid + 1];
  const float4 adv = *(const float4*)(ad_ + (size_t)wid * 4);
  // ---- online softmax statistics (one gather pass) -------------------------
  float m0 = -1e30f, m1 = -1e30f, m2 = -1e30f, m3 = -1e30f;
  float d0 = 0.f, d1 = 0.f, d2 = 0.f, d3 = 0.f;
  for (int i = s0 + lane; i < s1; i += 32) {
    const float4 av = *(const float4*)(as_ + (size_t)col[i] * 4);
    float v, M;
    v = lrelu(av.x + adv.x); M = fmaxf(m0, v);
    d0 = d0 * __expf(m0 - M) + __expf(v - M); m0 = M;
    v = lrelu(av.y + adv.y); M = fmaxf(m1, v);
    d1 = d1 * __expf(m1 - M) + __expf(v - M); m1 = M;
    v = lrelu(av.z + adv.z); M = fmaxf(m2, v);
    d2 = d2 * __expf(m2 - M) + __expf(v - M); m2 = M;
    v = lrelu(av.w + adv.w); M = fmaxf(m3, v);
    d3 = d3 * __expf(m3 - M) + __expf(v - M); m3 = M;
  }
#pragma unroll
  for (int off = 16; off >= 1; off >>= 1) {
    float mo, dd, M;
    mo = __shfl_xor(m0, off, 32); dd = __shfl_xor(d0, off, 32);
    M = fmaxf(m0, mo); d0 = d0 * __expf(m0 - M) + dd * __expf(mo - M); m0 = M;
    mo = __shfl_xor(m1, off, 32); dd = __shfl_xor(d1, off, 32);
    M = fmaxf(m1, mo); d1 = d1 * __expf(m1 - M) + dd * __expf(mo - M); m1 = M;
    mo = __shfl_xor(m2, off, 32); dd = __shfl_xor(d2, off, 32);
    M = fmaxf(m2, mo); d2 = d2 * __expf(m2 - M) + dd * __expf(mo - M); m2 = M;
    mo = __shfl_xor(m3, off, 32); dd = __shfl_xor(d3, off, 32);
    M = fmaxf(m3, mo); d3 = d3 * __expf(m3 - M) + dd * __expf(mo - M); m3 = M;
  }
  // ---- weighted message accumulation ---------------------------------------
  const int   c0  = lane * 2;         // 2 channels per lane, same head
  const int   hd  = c0 >> 4;
  const float mh  = hd == 0 ? m0 : hd == 1 ? m1 : hd == 2 ? m2 : m3;
  const float dh  = hd == 0 ? d0 : hd == 1 ? d1 : hd == 2 ? d2 : d3;
  const float adh = hd == 0 ? adv.x : hd == 1 ? adv.y : hd == 2 ? adv.z : adv.w;
  const float inv = 1.0f / (dh + 1e-16f);
  float a0 = 0.f, a1 = 0.f;
  for (int base = s0; base < s1; base += 32) {
    const int rem = s1 - base;
    int sv = 0;
    if (lane < rem) {
      sv = col[base + lane];                          // coalesced id load
      __builtin_prefetch(h + (size_t)sv * HID, 0, 0); // warm the 32 source rows
    }
    const int cnt = rem < 32 ? rem : 32;
    for (int j = 0; j < cnt; ++j) {
      const int s = __shfl(sv, j, 32);                // broadcast id from regs
      const float w = __expf(lrelu(as_[s * 4 + hd] + adh) - mh) * inv;
      const float2 hv = *(const float2*)(h + (size_t)s * HID + c0);
      a0 = fmaf(hv.x, w, a0);
      a1 = fmaf(hv.y, w, a1);
    }
  }
  out[(size_t)wid * HID + c0]     = fmaxf(a0 + bias[c0],     0.f);
  out[(size_t)wid * HID + c0 + 1] = fmaxf(a1 + bias[c0 + 1], 0.f);
}

// ---------------------------------------------------------------------------
// Global max pool over sorted batch ids (post-ReLU values >= 0, so int-bit
// atomicMax matches float ordering; few atomics thanks to per-thread runs).
// ---------------------------------------------------------------------------
__global__ void pool_k(const float* __restrict__ feat, const int* __restrict__ batch,
                       float* __restrict__ g, int N) {
  int tid = blockIdx.x * blockDim.x + threadIdx.x;
  int c = tid & 63;
  int n0 = (tid >> 6) * 64;
  if (n0 >= N) return;
  int n1 = n0 + 64;
  if (n1 > N) n1 = N;
  int cg = batch[n0];
  float run = 0.0f;
  for (int n = n0; n < n1; ++n) {
    int bg = batch[n];
    if (bg != cg) {
      atomicMax((int*)&g[cg * HID + c], __float_as_int(run));
      run = 0.0f;
      cg = bg;
    }
    run = fmaxf(run, feat[(size_t)n * HID + c]);
  }
  atomicMax((int*)&g[cg * HID + c], __float_as_int(run));
}

// ---------------------------------------------------------------------------
// Tiny head MLP: [64,64] -> relu([64,32]) -> [64,10], one block.
// ---------------------------------------------------------------------------
__global__ void mlp_k(const float* __restrict__ g, const float* __restrict__ W1,
                      const float* __restrict__ b1, const float* __restrict__ W2,
                      const float* __restrict__ b2, float* __restrict__ out) {
  __shared__ float t[64 * 32];
  const int tid = threadIdx.x;
  for (int idx = tid; idx < 64 * 32; idx += blockDim.x) {
    int gi = idx >> 5, j = idx & 31;
    float sum = b1[j];
#pragma unroll
    for (int k = 0; k < 64; ++k) sum = fmaf(g[gi * 64 + k], W1[k * 32 + j], sum);
    t[idx] = fmaxf(sum, 0.0f);
  }
  __syncthreads();
  for (int idx = tid; idx < 64 * 10; idx += blockDim.x) {
    int gi = idx / 10, cl = idx % 10;
    float sum = b2[cl];
#pragma unroll
    for (int j = 0; j < 32; ++j) sum = fmaf(t[gi * 32 + j], W2[j * 10 + cl], sum);
    out[idx] = sum;
  }
}

// ---------------------------------------------------------------------------
extern "C" void kernel_launch(void* const* d_in, const int* in_sizes, int n_in,
                              void* d_out, int out_size, void* d_ws, size_t ws_size,
                              hipStream_t stream) {
  (void)n_in; (void)out_size; (void)ws_size;
  const float* x       = (const float*)d_in[0];
  const int*   ei      = (const int*)d_in[1];
  const int*   batch   = (const int*)d_in[2];
  const float* W_in    = (const float*)d_in[3];
  const float* asrc_in = (const float*)d_in[4];
  const float* adst_in = (const float*)d_in[5];
  const float* b_in    = (const float*)d_in[6];
  const float* W_hid   = (const float*)d_in[7];
  const float* asrc_h  = (const float*)d_in[8];
  const float* adst_h  = (const float*)d_in[9];
  const float* b_hid   = (const float*)d_in[10];
  const float* W1      = (const float*)d_in[11];
  const float* b1      = (const float*)d_in[12];
  const float* W2      = (const float*)d_in[13];
  const float* b2      = (const float*)d_in[14];
  float* out = (float*)d_out;

  const int N  = in_sizes[0] / FIN;   // 100000
  const int E  = in_sizes[1] / 2;     // 1600000
  const int Et = E + N;               // + self-loops

  size_t off = 0;
  auto carve = [&](size_t bytes) -> char* {
    char* p = (char*)d_ws + off;
    off += (bytes + 255) & ~(size_t)255;
    return p;
  };
  float*    h       = (float*)carve((size_t)N * HID * 4);
  float*    feat    = (float*)carve((size_t)N * HID * 4);
  float*    as_     = (float*)carve((size_t)N * HEADS * 4);
  float*    ad_     = (float*)carve((size_t)N * HEADS * 4);
  int*      row_ptr = (int*)carve((size_t)(N + 1) * 4);
  int*      cursor  = (int*)carve((size_t)N * 4);
  int*      col     = (int*)carve((size_t)Et * 4);
  _Float16* Wt_hi   = (_Float16*)carve((size_t)FIN * HID * 2);
  _Float16* Wt_lo   = (_Float16*)carve((size_t)FIN * HID * 2);
  float*    g       = (float*)carve((size_t)64 * HID * 4);

  const int B = 256;
  // ---- CSR build (dst-sorted, reused by all 7 layers) ----------------------
  zero_i32<<<(N + B - 1) / B, B, 0, stream>>>(cursor, N);
  edge_count<<<(Et + B - 1) / B, B, 0, stream>>>(ei + E, cursor, E, N);
  scan_excl<<<1, 1024, 0, stream>>>(cursor, row_ptr, N);
  copy_i32<<<(N + B - 1) / B, B, 0, stream>>>(row_ptr, cursor, N);
  edge_scatter<<<(Et + B - 1) / B, B, 0, stream>>>(ei, ei + E, cursor, col, E, N);

  const int tiles      = (N + 15) / 16;
  const int gemm_blks  = (tiles + 7) / 8;
  const int alpha_blks = (N * HEADS + B - 1) / B;
  const int agg_blks   = (N * 32 + B - 1) / B;

  for (int l = 0; l < 1 + NLAYERS; ++l) {
    const int    K   = (l == 0) ? FIN : HID;
    const float* Wl  = (l == 0) ? W_in    : W_hid  + (size_t)(l - 1) * HID * HID;
    const float* asl = (l == 0) ? asrc_in : asrc_h + (size_t)(l - 1) * HEADS * CDIM;
    const float* adl = (l == 0) ? adst_in : adst_h + (size_t)(l - 1) * HEADS * CDIM;
    const float* bl  = (l == 0) ? b_in    : b_hid  + (size_t)(l - 1) * HID;
    const float* fin = (l == 0) ? x : feat;
    wcvt<<<(K * HID + B - 1) / B, B, 0, stream>>>(Wl, Wt_hi, Wt_lo, K);
    gemm_wmma_split<<<gemm_blks, B, 0, stream>>>(fin, Wt_hi, Wt_lo, h, N, K);
    alpha_k<<<alpha_blks, B, 0, stream>>>(h, asl, adl, as_, ad_, N);
    gat_aggregate<<<agg_blks, B, 0, stream>>>(h, as_, ad_, bl, row_ptr, col, feat, N);
  }

  zero_i32<<<(64 * HID + B - 1) / B, B, 0, stream>>>((int*)g, 64 * HID);
  const int segs = (N + 63) / 64;
  pool_k<<<(segs * 64 + B - 1) / B, B, 0, stream>>>(feat, batch, g, N);
  mlp_k<<<1, 256, 0, stream>>>(g, W1, b1, W2, b2, out);
}
